// UNet3LMCD_6451040878763
// MI455X (gfx1250) — matrix operand order repaired and checked by reading, not compiled, not run on hardware
//
#include <hip/hip_runtime.h>
#include <hip/hip_bf16.h>

typedef __attribute__((ext_vector_type(16))) _Float16 v16h;
typedef __attribute__((ext_vector_type(8)))  _Float16 v8h;
typedef __attribute__((ext_vector_type(8)))  float    v8f;

// ---------------- f32 -> f16 convert ----------------
__global__ void k_f32_to_f16(const float* __restrict__ src, _Float16* __restrict__ dst, long n) {
    long i = (long)blockIdx.x * blockDim.x + threadIdx.x;
    if (i < n) dst[i] = (_Float16)src[i];
}

// ---------------- zero the 256-float stats buffer ----------------
__global__ void k_zero256(float* __restrict__ p) {
    p[threadIdx.x] = 0.0f;
}

// ---------------- pack W[27,Cin,Cout] (f32) into WMMA B-fragments (f16) ----
// Layout: Bp[(((k*KC + kc)*NT + nt)*32 + lane)*16 + 2v + h]
//   lane 0-15 : N = lane,    K(chunk-rel) = s<8 ? s : 16 + (s-8)      (s=2v+h)
//   lane 16-31: N = lane-16, K(chunk-rel) = s<8 ? 8 + s : 24 + (s-8)
__global__ void k_pack_w(const float* __restrict__ W, int Cin, int Cout,
                         _Float16* __restrict__ Bp, int KC, int NT) {
    long t = (long)blockIdx.x * blockDim.x + threadIdx.x;
    long total = 27L * KC * NT * 32;
    if (t >= total) return;
    int  lane = (int)(t & 31);
    long u    = t >> 5;
    int  nt   = (int)(u % NT); u /= NT;
    int  kc   = (int)(u % KC);
    int  k    = (int)(u / KC);
    int  n    = nt * 16 + (lane & 15);
    int  g    = lane >> 4;
    _Float16* out = Bp + t * 16;
#pragma unroll
    for (int v = 0; v < 8; ++v) {
#pragma unroll
        for (int h = 0; h < 2; ++h) {
            int   kr = kc * 32 + g * 8 + ((2 * v + h) < 8 ? (2 * v + h) : 8 + (2 * v + h));
            float w  = (kr < Cin && n < Cout) ? W[((long)k * Cin + kr) * Cout + n] : 0.0f;
            out[2 * v + h] = (_Float16)w;
        }
    }
}

// ---------------- sparse conv: gather + WMMA f32<-f16 ----------------
// One wave computes a 16(rows) x 16*NPER(cols) tile of z[Nout, ldz].
// CIN compile-time => kc loop fully unrolled; bounds checks fold to constants.
template <int CIN, int NPER>
__global__ void __launch_bounds__(32)
k_sconv_wmma(const _Float16* __restrict__ x, int ldx,
             const _Float16* __restrict__ Bp, int NT,
             const int* __restrict__ map, int Nout,
             float* __restrict__ z, int ldz, int CoutStore) {
    constexpr int KC = (CIN + 31) / 32;
    const int lane = threadIdx.x;         // 0..31
    const int m16  = lane & 15;
    const int hi   = lane >> 4;           // 0 | 1
    const int tile = blockIdx.x;
    const int nt0  = blockIdx.y * NPER;   // first N-tile of this wave
    const int j    = tile * 16 + m16;     // output row this lane gathers for
    const int jc   = (j < Nout) ? j : (Nout - 1);

    v8f acc[NPER];
#pragma unroll
    for (int q = 0; q < NPER; ++q) acc[q] = {};

    for (int k = 0; k < 27; ++k) {
        int row = (j < Nout) ? map[(long)k * Nout + j] : -1;
        if (k < 26) __builtin_prefetch((const void*)(map + (long)(k + 1) * Nout + jc), 0, 1);
        const _Float16* xr = x + (long)row * ldx;

#pragma unroll
        for (int kc = 0; kc < KC; ++kc) {
            // ---- A fragment: 16x32 f16; lane holds K = base0..+7 and base0+16..+23
            v16h a = {};
            if (row >= 0) {
                const int base0 = kc * 32 + hi * 8;       // group 0 start
                const int base1 = base0 + 16;             // group 1 start
                // group g fully in-bounds iff kc*32 + g*16 + 16 <= CIN (covers both hi);
                // conditions are constant per unrolled kc iteration.
                if (kc * 32 + 16 <= CIN) {
                    v8h lo = *(const v8h*)(xr + base0);
#pragma unroll
                    for (int i = 0; i < 8; ++i) a[i] = lo[i];
                } else if (kc * 32 < CIN) {
#pragma unroll
                    for (int i = 0; i < 8; ++i)
                        if (base0 + i < CIN) a[i] = xr[base0 + i];
                }
                if (kc * 32 + 32 <= CIN) {
                    v8h hh = *(const v8h*)(xr + base1);
#pragma unroll
                    for (int i = 0; i < 8; ++i) a[8 + i] = hh[i];
                } else if (kc * 32 + 16 < CIN) {
#pragma unroll
                    for (int i = 0; i < 8; ++i)
                        if (base1 + i < CIN) a[8 + i] = xr[base1 + i];
                }
            }
            // ---- B fragments (prepacked, contiguous 32B/lane), one WMMA per N-tile
#pragma unroll
            for (int q = 0; q < NPER; ++q) {
                const v16h bfrag = *(const v16h*)(
                    Bp + ((((long)k * KC + kc) * NT + (nt0 + q)) * 32 + lane) * 16);
                acc[q] = __builtin_amdgcn_wmma_f32_16x16x32_f16(
                    false, a, false, bfrag, (short)0, acc[q], false, false);
            }
        }
    }

    // ---- store D: lane N = m16, VGPR r -> M = r + 8*hi
#pragma unroll
    for (int q = 0; q < NPER; ++q) {
        int col = (nt0 + q) * 16 + m16;
        if (col < CoutStore) {
#pragma unroll
            for (int r = 0; r < 8; ++r) {
                int jr = tile * 16 + r + hi * 8;
                if (jr < Nout) z[(long)jr * ldz + col] = acc[q][r];
            }
        }
    }
}

// ---------------- BN stats: per-channel sum / sumsq ----------------
// 256 threads, C in {32,64,128} (power of two), 128 rows per block.
__global__ void k_bn_stats(const float* __restrict__ z, int N, int C,
                           float* __restrict__ gsum, float* __restrict__ gsq) {
    __shared__ float ssum[256];
    __shared__ float ssq[256];
    int  tid  = threadIdx.x;
    int  c    = tid & (C - 1);
    int  jper = 256 / C;
    long r0   = (long)blockIdx.x * 128;
    float s = 0.0f, q = 0.0f;
    for (int jj = tid / C; jj < 128; jj += jper) {
        long r = r0 + jj;
        if (r < N) {
            float v = z[r * C + c];
            s += v; q += v * v;
        }
    }
    ssum[tid] = s; ssq[tid] = q;
    __syncthreads();
    for (int st = 128; st >= C; st >>= 1) {
        if (tid < st) { ssum[tid] += ssum[tid + st]; ssq[tid] += ssq[tid + st]; }
        __syncthreads();
    }
    if (tid < C) {
        atomicAdd(&gsum[tid], ssum[tid]);
        atomicAdd(&gsq[tid],  ssq[tid]);
    }
}

// ---------------- fused BN + ReLU + f16 convert + (concat) store ----------
__global__ void k_bn_relu_f16(const float* __restrict__ z, int N, int C,
                              const float* __restrict__ gsum, const float* __restrict__ gsq,
                              const float* __restrict__ g, const float* __restrict__ b,
                              _Float16* __restrict__ dst, int ldd, int colOff) {
    long i = (long)blockIdx.x * blockDim.x + threadIdx.x;
    if (i >= (long)N * C) return;
    int  c    = (int)(i % C);
    long r    = i / C;
    float invN = 1.0f / (float)N;
    float mu   = gsum[c] * invN;
    float var  = gsq[c] * invN - mu * mu;
    float sc   = g[c] * rsqrtf(var + 1e-5f);
    float y    = (z[i] - mu) * sc + b[c];
    y = y > 0.0f ? y : 0.0f;
    dst[r * ldd + colOff + c] = (_Float16)y;
}

extern "C" void kernel_launch(void* const* d_in, const int* in_sizes, int n_in,
                              void* d_out, int out_size, void* d_ws, size_t ws_size,
                              hipStream_t stream) {
    const float* feats = (const float*)d_in[0];
    const float* W0  = (const float*)d_in[1];
    const float* g0  = (const float*)d_in[2];
    const float* b0  = (const float*)d_in[3];
    const float* W1  = (const float*)d_in[4];
    const float* g1  = (const float*)d_in[5];
    const float* b1  = (const float*)d_in[6];
    const float* W2  = (const float*)d_in[7];
    const float* g2  = (const float*)d_in[8];
    const float* b2  = (const float*)d_in[9];
    const float* W2t = (const float*)d_in[10];
    const float* g2t = (const float*)d_in[11];
    const float* b2t = (const float*)d_in[12];
    const float* W1t = (const float*)d_in[13];
    const float* g1t = (const float*)d_in[14];
    const float* b1t = (const float*)d_in[15];
    const float* W0t = (const float*)d_in[16];
    const int* map0  = (const int*)d_in[17];
    const int* map1  = (const int*)d_in[18];
    const int* map2  = (const int*)d_in[19];
    const int* map2t = (const int*)d_in[20];
    const int* map1t = (const int*)d_in[21];
    const int* map0t = (const int*)d_in[22];
    (void)n_in; (void)out_size; (void)ws_size;

    const int N0 = in_sizes[0] / 16;
    const int N1 = in_sizes[18] / 27;
    const int N2 = in_sizes[19] / 27;

    // ---- carve workspace ----
    char* wsp = (char*)d_ws;
    auto carve = [&](size_t bytes) -> void* {
        void* p = (void*)wsp;
        wsp += (bytes + 255) & ~(size_t)255;
        return p;
    };
    _Float16* x0    = (_Float16*)carve((size_t)N0 * 16  * sizeof(_Float16));
    _Float16* s1cat = (_Float16*)carve((size_t)N0 * 96  * sizeof(_Float16)); // [d1 | s1]
    _Float16* s2cat = (_Float16*)carve((size_t)N1 * 128 * sizeof(_Float16)); // [d2 | s2]
    _Float16* s4    = (_Float16*)carve((size_t)N2 * 128 * sizeof(_Float16));
    size_t zel = (size_t)N0 * 64;
    if ((size_t)N1 * 64  > zel) zel = (size_t)N1 * 64;
    if ((size_t)N2 * 128 > zel) zel = (size_t)N2 * 128;
    if ((size_t)N0 * 32  > zel) zel = (size_t)N0 * 32;
    float* z     = (float*)carve(zel * sizeof(float));
    float* stats = (float*)carve(256 * sizeof(float));
    float* gsum  = stats;
    float* gsq   = stats + 128;
    _Float16* Wp0  = (_Float16*)carve(27L * 1 * 2 * 32 * 16 * sizeof(_Float16));
    _Float16* Wp1  = (_Float16*)carve(27L * 1 * 4 * 32 * 16 * sizeof(_Float16));
    _Float16* Wp2  = (_Float16*)carve(27L * 2 * 8 * 32 * 16 * sizeof(_Float16));
    _Float16* Wp2t = (_Float16*)carve(27L * 4 * 4 * 32 * 16 * sizeof(_Float16));
    _Float16* Wp1t = (_Float16*)carve(27L * 4 * 4 * 32 * 16 * sizeof(_Float16));
    _Float16* Wp0t = (_Float16*)carve(27L * 3 * 1 * 32 * 16 * sizeof(_Float16));

    // ---- convert input features to f16 ----
    {
        long n = (long)N0 * 16;
        k_f32_to_f16<<<dim3((unsigned)((n + 255) / 256)), dim3(256), 0, stream>>>(feats, x0, n);
    }

    // ---- prepack all weights ----
    auto pack = [&](const float* W, int Cin, int Cout, _Float16* Bp, int KC, int NT) {
        long total = 27L * KC * NT * 32;
        k_pack_w<<<dim3((unsigned)((total + 255) / 256)), dim3(256), 0, stream>>>(W, Cin, Cout, Bp, KC, NT);
    };
    pack(W0,  16,  32,  Wp0,  1, 2);
    pack(W1,  32,  64,  Wp1,  1, 4);
    pack(W2,  64,  128, Wp2,  2, 8);
    pack(W2t, 128, 64,  Wp2t, 4, 4);
    pack(W1t, 128, 64,  Wp1t, 4, 4);
    pack(W0t, 96,  2,   Wp0t, 3, 1);

    auto bn = [&](const float* zi, int N, int C, const float* g, const float* b,
                  _Float16* dst, int ldd, int colOff) {
        k_zero256<<<dim3(1), dim3(256), 0, stream>>>(stats);
        k_bn_stats<<<dim3((unsigned)((N + 127) / 128)), dim3(256), 0, stream>>>(zi, N, C, gsum, gsq);
        long total = (long)N * C;
        k_bn_relu_f16<<<dim3((unsigned)((total + 255) / 256)), dim3(256), 0, stream>>>(
            zi, N, C, gsum, gsq, g, b, dst, ldd, colOff);
    };

    // ---- encoder ----
    // conv0: Cin=16, NT=2, NPER=2 -> grid.y = 1
    k_sconv_wmma<16, 2><<<dim3((unsigned)((N0 + 15) / 16), 1), dim3(32), 0, stream>>>(
        x0, 16, Wp0, 2, map0, N0, z, 32, 32);
    bn(z, N0, 32, g0, b0, s1cat, 96, 64);                        // s1 -> s1cat[:,64:96]

    // conv1: Cin=32, NT=4, NPER=4 -> grid.y = 1
    k_sconv_wmma<32, 4><<<dim3((unsigned)((N1 + 15) / 16), 1), dim3(32), 0, stream>>>(
        s1cat + 64, 96, Wp1, 4, map1, N1, z, 64, 64);
    bn(z, N1, 64, g1, b1, s2cat, 128, 64);                       // s2 -> s2cat[:,64:128]

    // conv2: Cin=64, NT=8, NPER=4 -> grid.y = 2
    k_sconv_wmma<64, 4><<<dim3((unsigned)((N2 + 15) / 16), 2), dim3(32), 0, stream>>>(
        s2cat + 64, 128, Wp2, 8, map2, N2, z, 128, 128);
    bn(z, N2, 128, g2, b2, s4, 128, 0);                          // s4

    // ---- decoder ----
    // conv2t: Cin=128, NT=4, NPER=4 -> grid.y = 1
    k_sconv_wmma<128, 4><<<dim3((unsigned)((N1 + 15) / 16), 1), dim3(32), 0, stream>>>(
        s4, 128, Wp2t, 4, map2t, N1, z, 64, 64);
    bn(z, N1, 64, g2t, b2t, s2cat, 128, 0);                      // d2 -> s2cat[:,0:64]

    // conv1t: Cin=128, NT=4, NPER=4 -> grid.y = 1
    k_sconv_wmma<128, 4><<<dim3((unsigned)((N0 + 15) / 16), 1), dim3(32), 0, stream>>>(
        s2cat, 128, Wp1t, 4, map1t, N0, z, 64, 64);
    bn(z, N0, 64, g1t, b1t, s1cat, 96, 0);                       // d1 -> s1cat[:,0:64]

    // conv0t: Cin=96, NT=1, NPER=1 (Cout=2 padded to 16)
    k_sconv_wmma<96, 1><<<dim3((unsigned)((N0 + 15) / 16), 1), dim3(32), 0, stream>>>(
        s1cat, 96, Wp0t, 1, map0t, N0, (float*)d_out, 2, 2);
}
